// MLP_Interpolate_12223476924901
// MI455X (gfx1250) — matrix operand order repaired and checked by reading, not compiled, not run on hardware
//
#include <hip/hip_runtime.h>

typedef __attribute__((ext_vector_type(16))) _Float16 v16h;
typedef __attribute__((ext_vector_type(8)))  float    v8f;

#define LDSP 65  // padded LDS row stride (floats) for the 16x64 T tile

static __device__ inline v16h splat16(_Float16 s) {
    v16h v;
#pragma unroll
    for (int i = 0; i < 16; ++i) v[i] = s;
    return v;
}

__global__ __launch_bounds__(256, 1) void mlp_interp_kernel(
    const float* __restrict__ x,    // [4,64,128,128]
    const float* __restrict__ W1,   // [66,64]
    const float* __restrict__ b1,   // [64]
    const float* __restrict__ W2,   // [64,3]
    const float* __restrict__ b2,   // [3]
    float* __restrict__ out,        // [4,3,512,512]
    int nstrips)                    // 4*128*8 = 4096
{
    __shared__ float Tlds[8][16 * LDSP];   // per-wave T tile (33,280 B total)

    const int lane = threadIdx.x & 31;
    const int wave = threadIdx.x >> 5;
    const int hi   = lane >> 4;            // lane group (0: lanes 0-15, 1: 16-31)
    const int lp   = lane & 15;

    float* T = Tlds[wave];

    // ---- Preload stage-1 B operands: W1[0:64][0:64] as 8 v16h register tiles.
    // B layout (32x16 f16): lane group hi holds K = e + 16*hi, N = lp.
    v16h b1op[2][4];
#pragma unroll
    for (int h = 0; h < 2; ++h)
#pragma unroll
        for (int nt = 0; nt < 4; ++nt)
#pragma unroll
            for (int e = 0; e < 16; ++e) {
                int k = h * 32 + e + 16 * hi;
                int n = nt * 16 + lp;
                b1op[h][nt][e] = (_Float16)W1[k * 64 + n];
            }

    // ---- Preload stage-2 A operands: W2^T zero-padded to 16x64.
    // A layout (16x32 f16): M = lp (channel), K = e + 16*hi (+32 per half).
    v16h a2[2];
#pragma unroll
    for (int h = 0; h < 2; ++h)
#pragma unroll
        for (int e = 0; e < 16; ++e) {
            int k = h * 32 + e + 16 * hi;
            a2[h][e] = (lp < 3) ? (_Float16)W2[k * 3 + lp] : (_Float16)0.0f;
        }

    // ---- Per-lane rel-coordinate weight basis (packed f16) for the 32 features
    // this lane feeds into the stage-2 B operand: F(j) = (j&15) + 16*hi + 32*(j>>4).
    v16h wyh[2], wxh[2];
#pragma unroll
    for (int j = 0; j < 32; ++j) {
        int f = (j & 15) + 16 * hi + 32 * (j >> 4);
        wyh[j >> 4][j & 15] = (_Float16)W1[64 * 64 + f];
        wxh[j >> 4][j & 15] = (_Float16)W1[65 * 64 + f];
    }

    // ---- Bias folding via WMMA C-matrix init.
    // Stage 1: C lane holds feature n = nt*16+lp in all 8 VGPRs -> splat b1[n].
    v8f c1init[4];
#pragma unroll
    for (int nt = 0; nt < 4; ++nt) {
        float bv = b1[nt * 16 + lp];
#pragma unroll
        for (int v = 0; v < 8; ++v) c1init[nt][v] = bv;
    }
    // Stage 2: D row (VGPR) = channel for lanes 0-15 -> C = {b2[0],b2[1],b2[2],0,...}.
    v8f c2init = {};
    c2init[0] = b2[0];
    c2init[1] = b2[1];
    c2init[2] = b2[2];

    const v16h zeroh = {};

    const int gwave = blockIdx.x * 8 + wave;
    const int nw    = gridDim.x * 8;

    for (int s = gwave; s < nstrips; s += nw) {
        const int b   = s >> 10;          // 1024 strips per batch
        const int r   = s & 1023;
        const int iy  = r >> 3;           // input row
        const int ix0 = (r & 7) << 4;     // first input col of strip

        // ---- Stage 1: load A = x features (16 pixels x 64 ch) into WMMA A layout.
        // A: M = lp (pixel), K = e + 16*hi (+32 for second half).
        const float* xb = x + (size_t)b * 64 * 128 * 128;
        const int ix = ix0 + lp;
        v16h a1[2];
#pragma unroll
        for (int h = 0; h < 2; ++h)
#pragma unroll
            for (int e = 0; e < 16; ++e) {
                int k = h * 32 + e + 16 * hi;
                a1[h][e] = (_Float16)xb[((size_t)k * 128 + iy) * 128 + ix];
            }

        // ---- 4 N-tiles of T = x_feat @ W1[0:64] + b1 (bias via C-init) -> LDS.
        // D layout: lane holds N = lp (feature col of tile), VGPR v -> pixel v + 8*hi.
#pragma unroll
        for (int nt = 0; nt < 4; ++nt) {
            v8f acc = c1init[nt];
            acc = __builtin_amdgcn_wmma_f32_16x16x32_f16(false, a1[0], false, b1op[0][nt],
                                                         (short)0, acc, false, false);
            acc = __builtin_amdgcn_wmma_f32_16x16x32_f16(false, a1[1], false, b1op[1][nt],
                                                         (short)0, acc, false, false);
#pragma unroll
            for (int v = 0; v < 8; ++v) {
                int pix = v + 8 * hi;
                T[pix * LDSP + nt * 16 + lp] = acc[v];
            }
        }
        asm volatile("s_wait_dscnt 0" ::: "memory");  // T stores -> T reads (same wave)

        // ---- Read this lane's transposed T slice ONCE per strip (position-
        // invariant), packed to f16: element j holds T[pixel=lp][F(j)].
        v16h tvh[2];
#pragma unroll
        for (int j = 0; j < 32; ++j) {
            int f = (j & 15) + 16 * hi + 32 * (j >> 4);
            tvh[j >> 4][j & 15] = (_Float16)T[lp * LDSP + f];
        }

        // ---- Stage 2: 16 sub-positions (ry, rx); per position only packed-f16
        // fma + relu, then D = W2^T x H^T + b2 -> rows 0..2 are output channels.
        const int oyb = iy << 2;
        const int oxb = ((ix0 + lp) << 2);
        float* ob = out + (size_t)b * 3 * 512 * 512;

#pragma unroll
        for (int ry = 0; ry < 4; ++ry) {
            const v16h ryh = splat16((_Float16)(0.5f * ry - 0.75f));
            const v16h ty0 = __builtin_elementwise_fma(ryh, wyh[0], tvh[0]);
            const v16h ty1 = __builtin_elementwise_fma(ryh, wyh[1], tvh[1]);
#pragma unroll
            for (int rx = 0; rx < 4; ++rx) {
                const v16h rxh = splat16((_Float16)(0.5f * rx - 0.75f));
                v16h bh0 = __builtin_elementwise_max(
                    __builtin_elementwise_fma(rxh, wxh[0], ty0), zeroh);
                v16h bh1 = __builtin_elementwise_max(
                    __builtin_elementwise_fma(rxh, wxh[1], ty1), zeroh);

                v8f acc = c2init;
                acc = __builtin_amdgcn_wmma_f32_16x16x32_f16(false, a2[0], false, bh0,
                                                             (short)0, acc, false, false);
                acc = __builtin_amdgcn_wmma_f32_16x16x32_f16(false, a2[1], false, bh1,
                                                             (short)0, acc, false, false);
                if (lane < 16) {  // lane = pixel; acc[c] = channel c + b2[c]
                    const int oy = oyb + ry;
                    const int ox = oxb + rx;
                    const size_t base = (size_t)oy * 512 + ox;
                    ob[base]                 = acc[0];
                    ob[base + 512 * 512]     = acc[1];
                    ob[base + 2 * 512 * 512] = acc[2];
                }
            }
        }
    }
}

extern "C" void kernel_launch(void* const* d_in, const int* in_sizes, int n_in,
                              void* d_out, int out_size, void* d_ws, size_t ws_size,
                              hipStream_t stream) {
    const float* x  = (const float*)d_in[0];
    const float* W1 = (const float*)d_in[1];
    const float* b1 = (const float*)d_in[2];
    const float* W2 = (const float*)d_in[3];
    const float* b2 = (const float*)d_in[4];
    float* out = (float*)d_out;

    const int nstrips = 4 * 128 * 8;   // B * H * (W/16)
    dim3 grid(256), block(256);        // 2048 wave32s, 2 strips per wave
    mlp_interp_kernel<<<grid, block, 0, stream>>>(x, W1, b1, W2, b2, out, nstrips);
}
